// QuantizedLinear_21784074125376
// MI455X (gfx1250) — matrix operand order repaired and checked by reading, not compiled, not run on hardware
//
#include <hip/hip_runtime.h>
#include <stdint.h>

typedef __attribute__((ext_vector_type(16))) _Float16 v16h;
typedef __attribute__((ext_vector_type(8)))  _Float16 v8h;
typedef __attribute__((ext_vector_type(2)))  _Float16 h2;
typedef __attribute__((ext_vector_type(8)))  float    v8f;
typedef __attribute__((ext_vector_type(4)))  float    f32x4;

// Problem constants
#define M_DIM 8192
#define N_DIM 11008
#define K_DIM 4096

// Tiling: block tile 128(M) x 256(N), 8 waves, each wave 64x64 (4x4 WMMA frags)
#define BM 128
#define BN 256
#define BK 32
#define LDK (BK + 8)   // padded half-stride: 80 bytes -> conflict-free banks
#define KT (K_DIM / BK)

// Exact packed int8 -> f16 via magic-number trick:
//   ub = b ^ 0x80 (= b + 128), f16 bits (0x6400 | ub) == 1024 + ub,
//   subtract 1152.0 -> b exactly.  5 VALU ops per 4 bytes.
__device__ __forceinline__ void cvt_i8x4_f16(uint32_t d, h2& lo, h2& hi) {
    const uint32_t ub = d ^ 0x80808080u;
    uint32_t p0 = __builtin_amdgcn_perm(0x64646464u, ub, 0x05010400u); // {64,b1,64,b0}
    uint32_t p1 = __builtin_amdgcn_perm(0x64646464u, ub, 0x07030602u); // {64,b3,64,b2}
    h2 a, b;
    __builtin_memcpy(&a, &p0, 4);
    __builtin_memcpy(&b, &p1, 4);
    const h2 bias = {(_Float16)1152.0f, (_Float16)1152.0f};
    lo = a - bias;
    hi = b - bias;
}

__global__ __launch_bounds__(256) void qlinear_wmma_f16(
    const float* __restrict__ X,
    const int8_t* __restrict__ W,
    const float* __restrict__ scalep,
    float* __restrict__ Y)
{
    __shared__ alignas(16) _Float16 As[2][BM][LDK];   // x tile (f16), [m][k]  20480 B
    __shared__ alignas(16) _Float16 Bs[2][BN][LDK];   // w tile (f16), [n][k]  40960 B

    const int tid  = threadIdx.x;
    const int lane = tid & 31;
    const int wave = tid >> 5;
    const int wm   = (wave & 1) * 64;    // wave's M offset inside block tile
    const int wn   = (wave >> 1) * 64;   // wave's N offset inside block tile

    const int tileN = blockIdx.x * BN;
    const int tileM = blockIdx.y * BM;

    // X loads: 2 threads per row, 16 f32 each
    const int xrow = tid >> 1;            // 0..127
    const int xcol = (tid & 1) * 16;      // 0 or 16
    const float*  xg = X + (size_t)(tileM + xrow) * K_DIM + xcol;
    // W loads: 1 thread per row, 32 int8 each
    const int8_t* wg = W + (size_t)(tileN + tid) * K_DIM;

    v8f acc[4][4] = {};

    f32x4 xr[4];
    int4  wr[2];

    auto load_global = [&](int kt) {
        const float* xp = xg + kt * BK;
        #pragma unroll
        for (int i = 0; i < 4; ++i)
            xr[i] = *(const f32x4*)(xp + i * 4);
        const int8_t* wp = wg + kt * BK;
        wr[0] = *(const int4*)(wp);
        wr[1] = *(const int4*)(wp + 16);
    };

    auto store_lds = [&](int buf) {
        // x: f32 -> f16 (v_cvt_pk_f16_f32)
        _Float16 xh[16];
        #pragma unroll
        for (int i = 0; i < 4; ++i) {
            xh[4*i + 0] = (_Float16)xr[i][0];
            xh[4*i + 1] = (_Float16)xr[i][1];
            xh[4*i + 2] = (_Float16)xr[i][2];
            xh[4*i + 3] = (_Float16)xr[i][3];
        }
        *(v8h*)&As[buf][xrow][xcol]     = *(const v8h*)&xh[0];
        *(v8h*)&As[buf][xrow][xcol + 8] = *(const v8h*)&xh[8];

        // w: packed int8 -> f16 (perm + pk_add)
        _Float16 wh[32];
        h2* o = (h2*)wh;
        const uint32_t* wd = (const uint32_t*)&wr[0];
        #pragma unroll
        for (int i = 0; i < 8; ++i)
            cvt_i8x4_f16(wd[i], o[2*i], o[2*i + 1]);
        #pragma unroll
        for (int i = 0; i < 4; ++i)
            *(v8h*)&Bs[buf][tid][8*i] = *(const v8h*)&wh[8*i];
    };

    auto compute = [&](int buf) {
        union Frag { v16h v; v8h h[2]; };

        // B fragments: 32x16 f16, lane = N col (lane&15),
        // 16 contiguous K halves at (lane>>4)*16
        const int bn  = lane & 15;
        const int bkb = (lane >> 4) * 16;
        Frag b[4];
        #pragma unroll
        for (int nf = 0; nf < 4; ++nf) {
            const int row = wn + nf * 16 + bn;
            b[nf].h[0] = *(const v8h*)&Bs[buf][row][bkb];
            b[nf].h[1] = *(const v8h*)&Bs[buf][row][bkb + 8];
        }

        // A fragments streamed one at a time: 16x32 f16, lane = M row (lane&15),
        // 16 contiguous K halves at kb = (lane>>4)*16
        const int am  = lane & 15;
        const int akb = (lane >> 4) * 16;
        #pragma unroll
        for (int mf = 0; mf < 4; ++mf) {
            Frag a;
            const int row = wm + mf * 16 + am;
            a.h[0] = *(const v8h*)&As[buf][row][akb];
            a.h[1] = *(const v8h*)&As[buf][row][akb + 8];
            #pragma unroll
            for (int nf = 0; nf < 4; ++nf)
                acc[mf][nf] = __builtin_amdgcn_wmma_f32_16x16x32_f16(
                    /*neg_a=*/false, a.v,
                    /*neg_b=*/false, b[nf].v,
                    /*c_mod=*/(short)0, acc[mf][nf],
                    /*reuse_a=*/false, /*reuse_b=*/false);
        }
    };

    // ---- pipelined K loop ----
    load_global(0);
    store_lds(0);
    __syncthreads();

    for (int kt = 1; kt < KT; ++kt) {
        if (kt + 1 < KT) {
            __builtin_prefetch(xg + (size_t)(kt + 1) * BK, 0, 0);
            __builtin_prefetch(wg + (size_t)(kt + 1) * BK, 0, 0);
        }
        load_global(kt);          // global -> regs for stage kt
        compute((kt - 1) & 1);    // WMMA on previous stage
        __syncthreads();
        store_lds(kt & 1);        // regs -> LDS
        __syncthreads();
    }
    compute((KT - 1) & 1);

    // ---- epilogue: fold scalar dequant scale, store f32 ----
    const float scale = *scalep;
    const int cn = lane & 15;
    const int cm = (lane >> 4) * 8;
    #pragma unroll
    for (int mf = 0; mf < 4; ++mf) {
        #pragma unroll
        for (int nf = 0; nf < 4; ++nf) {
            const int col = tileN + wn + nf * 16 + cn;
            #pragma unroll
            for (int r = 0; r < 8; ++r) {
                const int row = tileM + wm + mf * 16 + cm + r;
                Y[(size_t)row * N_DIM + col] = acc[mf][nf][r] * scale;
            }
        }
    }
}

extern "C" void kernel_launch(void* const* d_in, const int* in_sizes, int n_in,
                              void* d_out, int out_size, void* d_ws, size_t ws_size,
                              hipStream_t stream) {
    const float*  x     = (const float*)d_in[0];
    const int8_t* w     = (const int8_t*)d_in[1];
    const float*  scale = (const float*)d_in[2];
    float*        out   = (float*)d_out;

    dim3 grid(N_DIM / BN, M_DIM / BM);   // (43, 64)
    dim3 block(256);
    qlinear_wmma_f16<<<grid, block, 0, stream>>>(x, w, scale, out);
}
